// DGATModule_47467978555681
// MI455X (gfx1250) — matrix operand order repaired and checked by reading, not compiled, not run on hardware
//
#include <hip/hip_runtime.h>
#include <hip/hip_bf16.h>
#include <math.h>

#define DIM    64
#define HEADS  8
#define HIDDEN 256

typedef __attribute__((ext_vector_type(16))) _Float16 v16h;
typedef __attribute__((ext_vector_type(8)))  _Float16 v8h;
typedef __attribute__((ext_vector_type(8)))  float    v8f;
typedef __attribute__((ext_vector_type(4)))  float    v4f;

union V16 { v16h v; v8h h[2]; _Float16 e[16]; };

static __device__ __forceinline__ v8f wmma_f16(v16h a, v16h b, v8f c) {
  // v_wmma_f32_16x16x32_f16: D = A(16x32 f16) * B(32x16 f16) + C(f32)
  return __builtin_amdgcn_wmma_f32_16x16x32_f16(false, a, false, b, (short)0, c,
                                                false, false);
}

// Non-returning relaxed agent-scope atomics -> single global_atomic_* ops
static __device__ __forceinline__ void atomic_add_f32(float* p, float v) {
  __hip_atomic_fetch_add(p, v, __ATOMIC_RELAXED, __HIP_MEMORY_SCOPE_AGENT);
}
static __device__ __forceinline__ void atomic_max_u32(unsigned* p, unsigned v) {
  __hip_atomic_fetch_max(p, v, __ATOMIC_RELAXED, __HIP_MEMORY_SCOPE_AGENT);
}

// ---- A fragment (16x32 f16) from row-major f32 source, converting on load.
// lane<16: M=lane, K = kb+0..7 and kb+16..23 ; lane>=16: K shifted by +8/+24.
static __device__ __forceinline__ v16h frag_a_f32(const float* __restrict__ base,
                                                  int stride, int r0, int kb, int lane) {
  const int row = r0 + (lane & 15);
  const int c0  = kb + 8 * (lane >> 4);
  const float* p = base + (size_t)row * stride + c0;
  const v4f a0 = *(const v4f*)(p);
  const v4f a1 = *(const v4f*)(p + 4);
  const v4f b0 = *(const v4f*)(p + 16);
  const v4f b1 = *(const v4f*)(p + 20);
  V16 u;
#pragma unroll
  for (int j = 0; j < 4; ++j) {
    u.e[j]      = (_Float16)a0[j];
    u.e[4 + j]  = (_Float16)a1[j];
    u.e[8 + j]  = (_Float16)b0[j];
    u.e[12 + j] = (_Float16)b1[j];
  }
  return u.v;
}

// ---- A fragment from row-major f16 source (global or LDS): two b128 loads.
static __device__ __forceinline__ v16h frag_a_f16(const _Float16* base, int stride,
                                                  int r0, int kb, int lane) {
  const int row = r0 + (lane & 15);
  const int c0  = kb + 8 * (lane >> 4);
  const _Float16* p = base + (size_t)row * stride + c0;
  V16 u;
  u.h[0] = *(const v8h*)(p);
  u.h[1] = *(const v8h*)(p + 16);
  return u.v;
}

// ---- B fragment (32x16 f16) from K-contiguous (pre-transposed) f16 weights.
// lane<16: N=lane, K=kb+0..7 and kb+16..23 ; lane>=16: N=lane-16, K shifted by +8/+24.
static __device__ __forceinline__ v16h frag_b_f16(const _Float16* base, int stride,
                                                  int n0, int kb, int lane) {
  const int n  = n0 + (lane & 15);
  const int c0 = kb + 8 * (lane >> 4);
  const _Float16* p = base + (size_t)n * stride + c0;
  V16 u;
  u.h[0] = *(const v8h*)(p);
  u.h[1] = *(const v8h*)(p + 16);
  return u.v;
}

// ---- A fragment of acc with per-head softmax normalization folded in.
// Column c head = c&7; c0 is a multiple of 8, so head index == j.
static __device__ __forceinline__ v16h frag_a_norm(const float* __restrict__ acc,
                                                   const float* __restrict__ denom,
                                                   int r0, int kb, int lane) {
  const int row = r0 + (lane & 15);
  const int c0  = kb + 8 * (lane >> 4);
  const v4f d0 = *(const v4f*)(denom + (size_t)row * HEADS);
  const v4f d1 = *(const v4f*)(denom + (size_t)row * HEADS + 4);
  float rd[8];
#pragma unroll
  for (int j = 0; j < 4; ++j) {
    rd[j]     = 1.0f / fmaxf(d0[j], 1e-12f);
    rd[4 + j] = 1.0f / fmaxf(d1[j], 1e-12f);
  }
  const float* p = acc + (size_t)row * DIM + c0;
  V16 u;
#pragma unroll
  for (int j = 0; j < 8; ++j) {
    u.e[j]     = (_Float16)(p[j]      * rd[j]);
    u.e[8 + j] = (_Float16)(p[16 + j] * rd[j]);
  }
  return u.v;
}

// ---- order-preserving float<->uint encoding for atomic max on f32 scores
static __device__ __forceinline__ unsigned enc_ord(float f) {
  unsigned b = __float_as_uint(f);
  return (b & 0x80000000u) ? ~b : (b | 0x80000000u);
}
static __device__ __forceinline__ float dec_ord(unsigned u) {
  unsigned b = (u & 0x80000000u) ? (u & 0x7FFFFFFFu) : ~u;
  return __uint_as_float(b);
}

// =====================================================================
// Kernel 0: weight prep — f16 transposes (K-contiguous) + collapsed edge path
// se = edge_feat @ (w_edge@w_att_e) + (b_edge@w_att_e + b_att_e)
// =====================================================================
__global__ void prep_kernel(const float* __restrict__ w_in,
                            const float* __restrict__ w_edge,
                            const float* __restrict__ b_edge,
                            const float* __restrict__ w_att_u,
                            const float* __restrict__ w_att_v,
                            const float* __restrict__ w_att_e,
                            const float* __restrict__ b_att_e,
                            const float* __restrict__ w_ff1,
                            const float* __restrict__ w_ff2,
                            _Float16* w_inT, _Float16* w_uvT,
                            _Float16* w_ff1T, _Float16* w_ff2T, float* wcomb) {
  const int t = threadIdx.x;  // 256 threads
  for (int idx = t; idx < 64 * 64; idx += 256) {
    const int n = idx >> 6, k = idx & 63;
    w_inT[n * 64 + k] = (_Float16)w_in[k * 64 + n];
  }
  for (int idx = t; idx < 16 * 64; idx += 256) {
    const int n = idx >> 6, k = idx & 63;
    const float v = (n < 8) ? w_att_u[k * 8 + n] : w_att_v[k * 8 + (n - 8)];
    w_uvT[n * 64 + k] = (_Float16)v;
  }
  for (int idx = t; idx < HIDDEN * 64; idx += 256) {
    const int n = idx >> 6, k = idx & 63;
    w_ff1T[n * 64 + k] = (_Float16)w_ff1[k * HIDDEN + n];
  }
  for (int idx = t; idx < 64 * HIDDEN; idx += 256) {
    const int n = idx >> 8, k = idx & 255;
    w_ff2T[n * HIDDEN + k] = (_Float16)w_ff2[k * 64 + n];
  }
  if (t < 8) {
    float c0 = 0.f, c1 = 0.f, cb = 0.f;
    for (int k = 0; k < 64; ++k) {
      const float we = w_att_e[k * 8 + t];
      c0 += w_edge[k] * we;
      c1 += w_edge[64 + k] * we;
      cb += b_edge[k] * we;
    }
    wcomb[t] = c0; wcomb[8 + t] = c1; wcomb[16 + t] = cb + b_att_e[t];
  }
}

// =====================================================================
// Kernel 1: re-init accumulators each call (graph replay safe)
// =====================================================================
__global__ void init_kernel(float* acc, float* denom, unsigned* mmax, int N) {
  const int i = blockIdx.x * blockDim.x + threadIdx.x;
  if (i < N * DIM) acc[i] = 0.0f;
  if (i < N * HEADS) { denom[i] = 0.0f; mmax[i] = 0u; }  // 0 decodes below all reals
}

// =====================================================================
// Kernel 2: h = x@w_in+b_in and fused [su|sv] = h@[w_att_u|w_att_v] (WMMA)
// one wave per 16-node tile; h bounced via LDS f16 to rebuild A fragments
// =====================================================================
__global__ void __launch_bounds__(128) node_linear_kernel(
    const float* __restrict__ x, const float* __restrict__ b_in,
    const float* __restrict__ b_att_u,
    const _Float16* __restrict__ w_inT, const _Float16* __restrict__ w_uvT,
    float* __restrict__ h, float* __restrict__ su_sv, int N) {
  __shared__ _Float16 hb[4][16 * DIM];  // 2KB per wave
  const int lane = threadIdx.x & 31;
  const int wv   = threadIdx.x >> 5;
  const int r0   = (blockIdx.x * 4 + wv) * 16;
  if (r0 >= N) return;

  const v16h a0 = frag_a_f32(x, DIM, r0, 0, lane);
  const v16h a1 = frag_a_f32(x, DIM, r0, 32, lane);
  _Float16* hrow = hb[wv];
  const int nsel = lane & 15;
  const int msel = 8 * (lane >> 4);

#pragma unroll
  for (int nt = 0; nt < 4; ++nt) {
    v8f c = {};
    c = wmma_f16(a0, frag_b_f16(w_inT, 64, nt * 16, 0, lane), c);
    c = wmma_f16(a1, frag_b_f16(w_inT, 64, nt * 16, 32, lane), c);
    const int col = nt * 16 + nsel;
    const float bias = b_in[col];
#pragma unroll
    for (int i = 0; i < 8; ++i) {
      const float v = c[i] + bias;
      h[(size_t)(r0 + i + msel) * DIM + col] = v;
      hrow[(i + msel) * DIM + col] = (_Float16)v;
    }
  }
  // su|sv = h_tile @ w_uvT  (cols 0-7 = su with bias, 8-15 = sv no bias)
  v8f cs = {};
  cs = wmma_f16(frag_a_f16(hrow, DIM, 0, 0, lane),  frag_b_f16(w_uvT, 64, 0, 0, lane),  cs);
  cs = wmma_f16(frag_a_f16(hrow, DIM, 0, 32, lane), frag_b_f16(w_uvT, 64, 0, 32, lane), cs);
  const float biasu = (nsel < 8) ? b_att_u[nsel] : 0.0f;
#pragma unroll
  for (int i = 0; i < 8; ++i)
    su_sv[(size_t)(r0 + i + msel) * 16 + nsel] = cs[i] + biasu;
}

// per-edge attention logits (recomputed in both edge passes; cheaper than
// storing the 51MB score tensor). Vectorized b128 gathers of su/sv rows.
static __device__ __forceinline__ void edge_scores(
    const float* __restrict__ su_sv, const float* __restrict__ wc,
    const float* __restrict__ ef, int s, int d, float* sc) {
  const float e0 = ef[0], e1 = ef[1];
  const v4f su0 = *(const v4f*)(su_sv + (size_t)s * 16);
  const v4f su1 = *(const v4f*)(su_sv + (size_t)s * 16 + 4);
  const v4f sv0 = *(const v4f*)(su_sv + (size_t)d * 16 + 8);
  const v4f sv1 = *(const v4f*)(su_sv + (size_t)d * 16 + 12);
#pragma unroll
  for (int j = 0; j < 4; ++j) {
    float v0 = su0[j] + sv0[j] + e0 * wc[j]     + e1 * wc[8 + j]     + wc[16 + j];
    float v1 = su1[j] + sv1[j] + e0 * wc[4 + j] + e1 * wc[12 + j]    + wc[20 + j];
    sc[j]     = (v0 > 0.0f) ? v0 : 0.2f * v0;   // LeakyReLU(0.2)
    sc[4 + j] = (v1 > 0.0f) ? v1 : 0.2f * v1;
  }
}

// =====================================================================
// Kernel 3: segment-max over dst (ordered-uint atomic max -> u32 hw atomic)
// =====================================================================
__global__ void edge_max_kernel(const int* __restrict__ src, const int* __restrict__ dst,
                                const float* __restrict__ edge_feat,
                                const float* __restrict__ su_sv,
                                const float* __restrict__ wc,
                                unsigned* __restrict__ mmax, int E) {
  const int e = blockIdx.x * blockDim.x + threadIdx.x;
  if (e >= E) return;
  const int s = src[e], d = dst[e];
  float sc[8];
  edge_scores(su_sv, wc, edge_feat + 2 * (size_t)e, s, d, sc);
#pragma unroll
  for (int hh = 0; hh < 8; ++hh)
    atomic_max_u32(&mmax[(size_t)d * HEADS + hh], enc_ord(sc[hh]));
}

// =====================================================================
// Kernel 4: fused exp + denom + unnormalized message scatter
// out[v] = (sum_e e_i * h_src) / denom[v]  — division deferred to FFN loader
// =====================================================================
__global__ void edge_accum_kernel(const int* __restrict__ src, const int* __restrict__ dst,
                                  const float* __restrict__ edge_feat,
                                  const float* __restrict__ su_sv,
                                  const float* __restrict__ wc,
                                  const unsigned* __restrict__ mmax,
                                  const float* __restrict__ h,
                                  float* __restrict__ denom, float* __restrict__ acc,
                                  int E) {
  const int e = blockIdx.x * blockDim.x + threadIdx.x;
  if (e >= E) return;
  const int s = src[e], d = dst[e];
  float sc[8];
  edge_scores(su_sv, wc, edge_feat + 2 * (size_t)e, s, d, sc);
  float w[8];
#pragma unroll
  for (int hh = 0; hh < 8; ++hh) {
    const float m = dec_ord(mmax[(size_t)d * HEADS + hh]);
    const float ev = __expf(sc[hh] - m);
    w[hh] = ev;
    atomic_add_f32(&denom[(size_t)d * HEADS + hh], ev);
  }
  const float* hs = h + (size_t)s * DIM;
  float* ad = acc + (size_t)d * DIM;
#pragma unroll
  for (int c4 = 0; c4 < 16; ++c4) {
    const v4f hv = *(const v4f*)(hs + c4 * 4);
#pragma unroll
    for (int j = 0; j < 4; ++j) {
      const int c = c4 * 4 + j;           // channel c belongs to head c&7
      atomic_add_f32(ad + c, hv[j] * w[c & 7]);
    }
  }
}

// =====================================================================
// Kernel 5: softmax-normalize + FFN (exact GELU), all GEMMs via WMMA.
// GEMM1 output staged as f16 in LDS so GEMM2 A-fragments are ds_load_b128.
// =====================================================================
__global__ void __launch_bounds__(128) ffn_kernel(
    const float* __restrict__ acc, const float* __restrict__ denom,
    const _Float16* __restrict__ w_ff1T, const _Float16* __restrict__ w_ff2T,
    const float* __restrict__ b_ff1, const float* __restrict__ b_ff2,
    float* __restrict__ out, int N) {
  __shared__ _Float16 tb[4][16 * HIDDEN];  // 8KB per wave
  const int lane = threadIdx.x & 31;
  const int wv   = threadIdx.x >> 5;
  const int r0   = (blockIdx.x * 4 + wv) * 16;
  if (r0 >= N) return;

  const v16h a0 = frag_a_norm(acc, denom, r0, 0, lane);
  const v16h a1 = frag_a_norm(acc, denom, r0, 32, lane);
  _Float16* t = tb[wv];
  const int nsel = lane & 15;
  const int msel = 8 * (lane >> 4);

  // GEMM1: [16,64] @ [64,256] + bias, exact GELU, f16 -> LDS
#pragma unroll
  for (int nt = 0; nt < 16; ++nt) {
    v8f c = {};
    c = wmma_f16(a0, frag_b_f16(w_ff1T, 64, nt * 16, 0, lane), c);
    c = wmma_f16(a1, frag_b_f16(w_ff1T, 64, nt * 16, 32, lane), c);
    const int col = nt * 16 + nsel;
    const float bias = b_ff1[col];
#pragma unroll
    for (int i = 0; i < 8; ++i) {
      float v = c[i] + bias;
      v = 0.5f * v * (1.0f + erff(v * 0.70710678f));  // exact GELU
      t[(i + msel) * HIDDEN + col] = (_Float16)v;
    }
  }

  // GEMM2: [16,256] @ [256,64] + bias -> out
#pragma unroll
  for (int nt = 0; nt < 4; ++nt) {
    v8f c = {};
#pragma unroll
    for (int k8 = 0; k8 < 8; ++k8)
      c = wmma_f16(frag_a_f16(t, HIDDEN, 0, k8 * 32, lane),
                   frag_b_f16(w_ff2T, HIDDEN, nt * 16, k8 * 32, lane), c);
    const int col = nt * 16 + nsel;
    const float bias = b_ff2[col];
#pragma unroll
    for (int i = 0; i < 8; ++i)
      out[(size_t)(r0 + i + msel) * DIM + col] = c[i] + bias;
  }
}

// =====================================================================
extern "C" void kernel_launch(void* const* d_in, const int* in_sizes, int n_in,
                              void* d_out, int out_size, void* d_ws, size_t ws_size,
                              hipStream_t stream) {
  const float* x         = (const float*)d_in[0];
  const float* edge_feat = (const float*)d_in[1];
  const float* w_in      = (const float*)d_in[2];
  const float* b_in      = (const float*)d_in[3];
  const float* w_edge    = (const float*)d_in[4];
  const float* b_edge    = (const float*)d_in[5];
  const float* w_att_u   = (const float*)d_in[6];
  const float* b_att_u   = (const float*)d_in[7];
  const float* w_att_v   = (const float*)d_in[8];
  const float* w_att_e   = (const float*)d_in[9];
  const float* b_att_e   = (const float*)d_in[10];
  const float* w_ff1     = (const float*)d_in[11];
  const float* b_ff1     = (const float*)d_in[12];
  const float* w_ff2     = (const float*)d_in[13];
  const float* b_ff2     = (const float*)d_in[14];
  const int*   src       = (const int*)d_in[15];
  const int*   dst       = (const int*)d_in[16];
  float* out = (float*)d_out;

  const int N = in_sizes[0] / DIM;
  const int E = in_sizes[15];

  // workspace carve-out (256B aligned)
  size_t off = 0;
  auto bump = [&](size_t bytes) -> void* {
    off = (off + 255) & ~(size_t)255;
    void* p = (char*)d_ws + off;
    off += bytes;
    return p;
  };
  float*     h      = (float*)bump((size_t)N * DIM * 4);
  float*     su_sv  = (float*)bump((size_t)N * 16 * 4);
  unsigned*  mmax   = (unsigned*)bump((size_t)N * HEADS * 4);
  float*     denom  = (float*)bump((size_t)N * HEADS * 4);
  float*     acc    = (float*)bump((size_t)N * DIM * 4);
  _Float16*  w_inT  = (_Float16*)bump((size_t)64 * 64 * 2);
  _Float16*  w_uvT  = (_Float16*)bump((size_t)16 * 64 * 2);
  _Float16*  w_ff1T = (_Float16*)bump((size_t)HIDDEN * 64 * 2);
  _Float16*  w_ff2T = (_Float16*)bump((size_t)64 * HIDDEN * 2);
  float*     wcomb  = (float*)bump(24 * 4);
  (void)ws_size; (void)n_in; (void)out_size;

  const int tiles      = (N + 15) / 16;
  const int nodeBlocks = (tiles + 3) / 4;
  const int edgeBlocks = (E + 255) / 256;
  const int initBlocks = (N * DIM + 255) / 256;

  prep_kernel<<<1, 256, 0, stream>>>(w_in, w_edge, b_edge, w_att_u, w_att_v,
                                     w_att_e, b_att_e, w_ff1, w_ff2,
                                     w_inT, w_uvT, w_ff1T, w_ff2T, wcomb);
  init_kernel<<<initBlocks, 256, 0, stream>>>(acc, denom, mmax, N);
  node_linear_kernel<<<nodeBlocks, 128, 0, stream>>>(x, b_in, b_att_u, w_inT,
                                                     w_uvT, h, su_sv, N);
  edge_max_kernel<<<edgeBlocks, 256, 0, stream>>>(src, dst, edge_feat, su_sv,
                                                  wcomb, mmax, E);
  edge_accum_kernel<<<edgeBlocks, 256, 0, stream>>>(src, dst, edge_feat, su_sv,
                                                    wcomb, mmax, h, denom, acc, E);
  ffn_kernel<<<nodeBlocks, 128, 0, stream>>>(acc, denom, w_ff1T, w_ff2T,
                                             b_ff1, b_ff2, out, N);
}
